// CompILESegmenter_28879360099143
// MI455X (gfx1250) — compile-verified, matrix operand order checked by reading, not run on hardware
//
#include <hip/hip_runtime.h>
#include <cstdint>
#include <cstddef>

// CDNA5 / gfx1250 implementation of the gated-GRU scan.
// Strategy: precompute gate + input-side GEMM (parallel, bf16 WMMA),
// run the sequential scan as one fused WMMA+pointwise kernel per step,
// defer mu/lv/z heads to one big WMMA GEMM at the end.
// Round 4: split-K(4) + LDS reduction in the latency-critical sequential
// step kernel: 4x shorter dependent load->wmma chain per step, 4x more
// waves (128 blocks) to cover the machine during the scan.

#define DEVFN __device__ __forceinline__

typedef __attribute__((ext_vector_type(16))) __bf16 bf16x16;
typedef __attribute__((ext_vector_type(8)))  __bf16 bf16x8;
typedef __attribute__((ext_vector_type(8)))  float  f32x8;

static constexpr int B_    = 64;
static constexpr int T_    = 512;
static constexpr int DIM_  = 1024;
static constexpr int ZDIM_ = 128;
static constexpr int BT_   = B_ * T_;
static constexpr int DIM3_ = 3 * DIM_;

// ---- WMMA helpers (layouts per CDNA5 ISA §7.12.2) -------------

// 16-bit A/B fragment 16x32: lane L holds row (L&15); lanes 0-15 cover
// K in {0..7,16..23}, lanes 16-31 cover K in {8..15,24..31} -> two
// contiguous 8-element (16B) chunks per lane.
DEVFN bf16x16 ld_frag32(const __bf16* p) {
  union { bf16x16 v; bf16x8 h[2]; } u;
  u.h[0] = *(const bf16x8*)(p);
  u.h[1] = *(const bf16x8*)(p + 16);
  return u.v;
}

DEVFN f32x8 wmma_bf16(bf16x16 a, bf16x16 b, f32x8 c) {
  return __builtin_amdgcn_wmma_f32_16x16x32_bf16(
      /*neg_a=*/false, a, /*neg_b=*/false, b,
      /*c_mod=*/(short)0, c, /*reuse_a=*/false, /*reuse_b=*/false);
}

DEVFN float sigmoidf_(float x) { return 1.0f / (1.0f + __expf(-x)); }

// K-loop over [k_begin, k_end): load clause then WMMA group per iteration.
template <int NF>
DEVFN void gemm_range(const __bf16* __restrict__ arow,
                      const __bf16* const (&wrow)[NF],
                      f32x8 (&acc)[NF], int kb, int k_begin, int k_end) {
  for (int k0 = k_begin; k0 < k_end; k0 += 32) {
    bf16x16 af = ld_frag32(arow + k0 + kb);
    bf16x16 bf[NF];
#pragma unroll
    for (int f = 0; f < NF; ++f) bf[f] = ld_frag32(wrow[f] + k0 + kb);
#pragma unroll
    for (int f = 0; f < NF; ++f) acc[f] = wmma_bf16(af, bf[f], acc[f]);
  }
}

// ---- utility kernels ------------------------------------------------------

__global__ void f32_to_bf16_kernel(const float* __restrict__ src,
                                   __bf16* __restrict__ dst, size_t n) {
  size_t i = (size_t)blockIdx.x * blockDim.x + threadIdx.x;
  size_t stride = (size_t)gridDim.x * blockDim.x;
  for (; i < n; i += stride) dst[i] = (__bf16)src[i];
}

__global__ void zero_state_kernel(float* __restrict__ sf,
                                  __bf16* __restrict__ sb, int n) {
  int i = blockIdx.x * blockDim.x + threadIdx.x;
  int stride = gridDim.x * blockDim.x;
  for (; i < n; i += stride) { sf[i] = 0.0f; sb[i] = (__bf16)0.0f; }
}

// ---- gate MLP (depends only on e_t -> fully parallel) ---------------------
// One wave32 per (b,t) row; lane j handles hidden units j and j+32.
// Kept in f32: the hard threshold g > 0.5 decides a full hidden-state reset,
// so we don't want bf16 rounding flipping borderline gates.
__global__ void gate_kernel(const float* __restrict__ e,
                            const float* __restrict__ w1,   // [64][1024]
                            const float* __restrict__ b1,   // [64]
                            const float* __restrict__ w2,   // [64]
                            const float* __restrict__ b2,   // [1]
                            float* __restrict__ g_out) {    // [BT]
  int r = blockIdx.x;
  int lane = threadIdx.x;
  const float* erow = e + (size_t)r * DIM_;
  const float* wa = w1 + (size_t)lane * DIM_;
  const float* wb = w1 + (size_t)(lane + 32) * DIM_;
  float a0 = 0.0f, a1 = 0.0f;
  for (int k = 0; k < DIM_; k += 4) {
    float4 ev = *(const float4*)(erow + k);
    float4 va = *(const float4*)(wa + k);
    float4 vb = *(const float4*)(wb + k);
    a0 += ev.x * va.x + ev.y * va.y + ev.z * va.z + ev.w * va.w;
    a1 += ev.x * vb.x + ev.y * vb.y + ev.z * vb.z + ev.w * vb.w;
  }
  float p = tanhf(a0 + b1[lane]) * w2[lane] +
            tanhf(a1 + b1[lane + 32]) * w2[lane + 32];
  for (int off = 16; off > 0; off >>= 1) p += __shfl_xor(p, off, 32);
  if (lane == 0) g_out[r] = sigmoidf_(p + b2[0]);
}

// ---- gi = e @ w_ih^T + b_ih : (32768x1024)x(1024x3072), bf16 WMMA ---------
// Each wave computes a 16x128 tile (8 accumulators). Throughput-bound;
// latency hidden by ~7 waves/SIMD occupancy.
__global__ void gi_gemm_kernel(const __bf16* __restrict__ A,     // [BT][1024]
                               const __bf16* __restrict__ W,     // [3072][1024]
                               const float* __restrict__ bias,   // [3072]
                               float* __restrict__ gi) {         // [BT][3072]
  const int NT = DIM3_ / 128;  // 24 n-tiles
  int wave = (blockIdx.x * blockDim.x + threadIdx.x) >> 5;
  int lane = threadIdx.x & 31;
  int m0 = (wave / NT) * 16;
  int n0 = (wave % NT) * 128;
  int lrow = lane & 15;
  int kb = (lane >> 4) << 3;

  f32x8 acc[8];
#pragma unroll
  for (int f = 0; f < 8; ++f) acc[f] = {};

  const __bf16* arow = A + (size_t)(m0 + lrow) * DIM_;
  const __bf16* wrow[8];
#pragma unroll
  for (int f = 0; f < 8; ++f)
    wrow[f] = W + (size_t)(n0 + f * 16 + lrow) * DIM_;

  gemm_range<8>(arow, wrow, acc, kb, 0, DIM_);

  int rowoff = (lane >> 4) << 3;  // C/D: M = r + 8 for lanes 16..31
#pragma unroll
  for (int f = 0; f < 8; ++f) {
    int col = n0 + f * 16 + (lane & 15);
    float bcol = bias[col];
#pragma unroll
    for (int r = 0; r < 8; ++r)
      gi[(size_t)(m0 + r + rowoff) * DIM3_ + col] = acc[f][r] + bcol;
  }
}

// ---- fused sequential step: split-K gh GEMM + LDS reduce + GRU update -----
// 256 output tiles (16 rows x 16 cols x {r,z,n} triple). Each tile is
// computed by 4 waves covering K-quarters (8 load->wmma iterations each,
// 4x shorter dependent chain than full K). Partials meet in LDS; the kq==0
// wave reduces and runs the pointwise GRU epilogue in-register.
__global__ void __launch_bounds__(256)
gru_step_kernel(const __bf16* __restrict__ s_bf,   // [64][1024]
                const float* __restrict__ s_f,     // [64][1024]
                const __bf16* __restrict__ Whh,    // [3072][1024]
                const float* __restrict__ b_hh,    // [3072]
                const float* __restrict__ gi,      // [BT][3072] bias folded
                const float* __restrict__ g,       // [BT] gate sigmoid
                float* __restrict__ s_f_next,
                __bf16* __restrict__ s_bf_next,
                __bf16* __restrict__ h_hist,       // [BT][1024]
                int t) {
  // smem: [2 tiles][3 quarters(kq=1..3)][3 frags][32 lanes][8 floats]
  __shared__ float smem[2 * 3 * 3 * 32 * 8];

  int lane = threadIdx.x & 31;
  int wv = threadIdx.x >> 5;        // 0..7
  int tileLocal = wv >> 2;          // 0..1
  int kq = wv & 3;                  // K quarter
  int tile = blockIdx.x * 2 + tileLocal;   // 0..255
  int m0 = (tile >> 6) << 4;        // 4 m-tiles (B = 64)
  int n0 = (tile & 63) << 4;        // 64 n-tiles (DIM = 1024)
  int lrow = lane & 15;
  int kb = (lane >> 4) << 3;

  f32x8 acc[3];
#pragma unroll
  for (int f = 0; f < 3; ++f) acc[f] = {};

  const __bf16* arow = s_bf + (size_t)(m0 + lrow) * DIM_;
  const __bf16* wrow[3];
#pragma unroll
  for (int f = 0; f < 3; ++f)
    wrow[f] = Whh + (size_t)(n0 + f * DIM_ + lrow) * DIM_;

  gemm_range<3>(arow, wrow, acc, kb, kq * (DIM_ / 4), (kq + 1) * (DIM_ / 4));

  // non-zero quarters publish partials to LDS
  if (kq != 0) {
#pragma unroll
    for (int f = 0; f < 3; ++f) {
      int base = ((((tileLocal * 3 + (kq - 1)) * 3 + f) * 32 + lane) * 8);
      *(f32x8*)(smem + base) = acc[f];
    }
  }
  __syncthreads();
  if (kq != 0) return;

  // kq==0 wave reduces the three partials
#pragma unroll
  for (int q = 0; q < 3; ++q) {
#pragma unroll
    for (int f = 0; f < 3; ++f) {
      int base = ((((tileLocal * 3 + q) * 3 + f) * 32 + lane) * 8);
      f32x8 p = *(const f32x8*)(smem + base);
#pragma unroll
      for (int r = 0; r < 8; ++r) acc[f][r] += p[r];
    }
  }

  int col = n0 + (lane & 15);
  float bh0 = b_hh[col], bh1 = b_hh[col + DIM_], bh2 = b_hh[col + 2 * DIM_];
  int rowoff = (lane >> 4) << 3;

#pragma unroll
  for (int r = 0; r < 8; ++r) {
    int m = m0 + r + rowoff;                    // batch index
    size_t girow = ((size_t)m * T_ + t) * DIM3_;
    float ir = gi[girow + col];
    float iz = gi[girow + DIM_ + col];
    float in_ = gi[girow + 2 * DIM_ + col];
    float rr = sigmoidf_(ir + acc[0][r] + bh0);
    float zg = sigmoidf_(iz + acc[1][r] + bh1);
    float nn = tanhf(in_ + rr * (acc[2][r] + bh2));
    float sp = s_f[(size_t)m * DIM_ + col];     // already reset-gated
    float h = (1.0f - zg) * nn + zg * sp;
    h_hist[((size_t)m * T_ + t) * DIM_ + col] = (__bf16)h;
    // fold next step's straight-through reset into the stored state
    float gnext = (t + 1 < T_) ? g[(size_t)m * T_ + (t + 1)] : 0.0f;
    float sn = (gnext > 0.5f) ? 0.0f : h;
    s_f_next[(size_t)m * DIM_ + col] = sn;
    s_bf_next[(size_t)m * DIM_ + col] = (__bf16)sn;
  }
}

// ---- heads: mu/lv GEMM + reparameterization (deferred, parallel) ----------
// Wave computes paired 16x16 tiles of mu (rows n0..) and lv (rows 128+n0..)
// so z = mu + exp(0.5 lv) * noise is formed in-register.
__global__ void heads_kernel(const __bf16* __restrict__ H,     // [BT][1024]
                             const __bf16* __restrict__ Wh,    // [256][1024] (mu;lv)
                             const float* __restrict__ mu_b,   // [128]
                             const float* __restrict__ lv_b,   // [128]
                             const float* __restrict__ noise,  // [BT][128]
                             float* __restrict__ out_z,
                             float* __restrict__ out_mu,
                             float* __restrict__ out_lv) {
  int wave = (blockIdx.x * blockDim.x + threadIdx.x) >> 5;
  int lane = threadIdx.x & 31;
  int m0 = (wave >> 3) << 4;   // 2048 m-tiles
  int n0 = (wave & 7) << 4;    // 8 n-tiles (ZDIM = 128)
  int lrow = lane & 15;
  int kb = (lane >> 4) << 3;

  f32x8 acc[2];
#pragma unroll
  for (int f = 0; f < 2; ++f) acc[f] = {};

  const __bf16* arow = H + (size_t)(m0 + lrow) * DIM_;
  const __bf16* wrow[2];
  wrow[0] = Wh + (size_t)(n0 + lrow) * DIM_;
  wrow[1] = Wh + (size_t)(ZDIM_ + n0 + lrow) * DIM_;

  gemm_range<2>(arow, wrow, acc, kb, 0, DIM_);

  int col = n0 + (lane & 15);
  float bm = mu_b[col], bl = lv_b[col];
  int rowoff = (lane >> 4) << 3;
#pragma unroll
  for (int r = 0; r < 8; ++r) {
    size_t row = (size_t)(m0 + r + rowoff);
    float mu = acc[0][r] + bm;
    float lv = acc[1][r] + bl;
    float z = mu + __expf(0.5f * lv) * noise[row * ZDIM_ + col];
    out_mu[row * ZDIM_ + col] = mu;
    out_lv[row * ZDIM_ + col] = lv;
    out_z[row * ZDIM_ + col] = z;
  }
}

// ---- host orchestration ---------------------------------------------------

extern "C" void kernel_launch(void* const* d_in, const int* in_sizes, int n_in,
                              void* d_out, int out_size, void* d_ws, size_t ws_size,
                              hipStream_t stream) {
  (void)in_sizes; (void)n_in; (void)out_size; (void)ws_size;
  const float* e       = (const float*)d_in[0];
  const float* noise   = (const float*)d_in[1];
  const float* gate_w1 = (const float*)d_in[2];
  const float* gate_b1 = (const float*)d_in[3];
  const float* gate_w2 = (const float*)d_in[4];
  const float* gate_b2 = (const float*)d_in[5];
  const float* w_ih    = (const float*)d_in[6];
  const float* b_ih    = (const float*)d_in[7];
  const float* w_hh    = (const float*)d_in[8];
  const float* b_hh    = (const float*)d_in[9];
  const float* mu_w    = (const float*)d_in[10];
  const float* mu_b    = (const float*)d_in[11];
  const float* lv_w    = (const float*)d_in[12];
  const float* lv_b    = (const float*)d_in[13];

  // d_out = [z | g | mu | lv] flat
  float* out   = (float*)d_out;
  float* out_z = out;
  float* out_g = out + (size_t)BT_ * ZDIM_;
  float* out_mu = out_g + BT_;
  float* out_lv = out_mu + (size_t)BT_ * ZDIM_;

  // workspace carve (~550 MB)
  size_t off = 0;
  auto carve = [&](size_t bytes) -> void* {
    void* p = (char*)d_ws + off;
    off += (bytes + 255) & ~(size_t)255;
    return p;
  };
  __bf16* whh_bf  = (__bf16*)carve((size_t)DIM3_ * DIM_ * 2);
  __bf16* wih_bf  = (__bf16*)carve((size_t)DIM3_ * DIM_ * 2);
  __bf16* head_bf = (__bf16*)carve((size_t)2 * ZDIM_ * DIM_ * 2);
  __bf16* e_bf    = (__bf16*)carve((size_t)BT_ * DIM_ * 2);
  float*  gi      = (float*) carve((size_t)BT_ * DIM3_ * 4);
  __bf16* h_hist  = (__bf16*)carve((size_t)BT_ * DIM_ * 2);
  float*  sA_f    = (float*) carve((size_t)B_ * DIM_ * 4);
  float*  sB_f    = (float*) carve((size_t)B_ * DIM_ * 4);
  __bf16* sA_bf   = (__bf16*)carve((size_t)B_ * DIM_ * 2);
  __bf16* sB_bf   = (__bf16*)carve((size_t)B_ * DIM_ * 2);

  // 1) precision conversion (weights + activations to bf16)
  f32_to_bf16_kernel<<<2048, 256, 0, stream>>>(w_hh, whh_bf, (size_t)DIM3_ * DIM_);
  f32_to_bf16_kernel<<<2048, 256, 0, stream>>>(w_ih, wih_bf, (size_t)DIM3_ * DIM_);
  f32_to_bf16_kernel<<<256, 256, 0, stream>>>(mu_w, head_bf, (size_t)ZDIM_ * DIM_);
  f32_to_bf16_kernel<<<256, 256, 0, stream>>>(lv_w, head_bf + (size_t)ZDIM_ * DIM_,
                                              (size_t)ZDIM_ * DIM_);
  f32_to_bf16_kernel<<<4096, 256, 0, stream>>>(e, e_bf, (size_t)BT_ * DIM_);

  // 2) h0 = 0 (ws is poisoned by the harness)
  zero_state_kernel<<<64, 256, 0, stream>>>(sA_f, sA_bf, B_ * DIM_);

  // 3) gate MLP for all (b,t)  — writes g directly to d_out
  gate_kernel<<<BT_, 32, 0, stream>>>(e, gate_w1, gate_b1, gate_w2, gate_b2, out_g);

  // 4) input-side GEMM for all (b,t): gi = e @ w_ih^T + b_ih
  gi_gemm_kernel<<<6144, 256, 0, stream>>>(e_bf, wih_bf, b_ih, gi);

  // 5) sequential scan: one fused split-K kernel per step
  for (int t = 0; t < T_; ++t) {
    const __bf16* sb = (t & 1) ? sB_bf : sA_bf;
    const float*  sf = (t & 1) ? sB_f  : sA_f;
    float*  sfn = (t & 1) ? sA_f  : sB_f;
    __bf16* sbn = (t & 1) ? sA_bf : sB_bf;
    gru_step_kernel<<<128, 256, 0, stream>>>(sb, sf, whh_bf, b_hh, gi, out_g,
                                             sfn, sbn, h_hist, t);
  }

  // 6) heads + reparameterization for all (b,t)
  heads_kernel<<<2048, 256, 0, stream>>>(h_hist, head_bf, mu_b, lv_b, noise,
                                         out_z, out_mu, out_lv);
}
